// SelfAttention_22600117911812
// MI455X (gfx1250) — matrix operand order repaired and checked by reading, not compiled, not run on hardware
//
#include <hip/hip_runtime.h>
#include <hip/hip_bf16.h>
#include <math.h>

// ---------------------------------------------------------------------------
// Self-attention block for MI455X (gfx1250, wave32, WMMA bf16 16x16x32)
// b=2, s=2048, d_model=2048, heads=16, d_k=128
// ---------------------------------------------------------------------------

typedef __attribute__((ext_vector_type(16))) __bf16       v16bf;
typedef __attribute__((ext_vector_type(8)))  float        v8f;
typedef __attribute__((ext_vector_type(4)))  unsigned int u32x4;

union FragBF { v16bf v; u32x4 q[2]; };

static __device__ __forceinline__ v8f vzero8() {
  v8f z;
  for (int i = 0; i < 8; ++i) z[i] = 0.f;
  return z;
}

static __device__ __forceinline__ v8f wmma_bf16(const FragBF& a, const FragBF& b, v8f c) {
  return __builtin_amdgcn_wmma_f32_16x16x32_bf16(false, a.v, false, b.v,
                                                 (short)0, c, false, false);
}

// async global->LDS copy of 16 bytes per lane (CDNA5 ASYNCcnt DMA path)
static __device__ __forceinline__ void async_copy_b128(const void* gaddr, void* lds) {
  const unsigned long long ga = (unsigned long long)(uintptr_t)gaddr;
  const unsigned lo = (unsigned)(uintptr_t)lds;   // low 32 bits = LDS offset
  asm volatile("global_load_async_to_lds_b128 %0, %1, off"
               :: "v"(lo), "v"(ga) : "memory");
}

static __device__ __forceinline__ void wait_async0() {
  asm volatile("s_wait_asynccnt 0x0" ::: "memory");
}
static __device__ __forceinline__ void wait_async4() {   // allow 4 in flight
  asm volatile("s_wait_asynccnt 0x4" ::: "memory");
}

// ------------------------------- fp32 -> bf16 weight convert ----------------
__global__ __launch_bounds__(256) void k_cvt(const float* __restrict__ in,
                                             __bf16* __restrict__ out) {
  int i = (blockIdx.x * 256 + threadIdx.x) * 8;     // exactly 4M elements
#pragma unroll
  for (int j = 0; j < 8; ++j) out[i + j] = (__bf16)in[i + j];
}

// ------------------------------- LayerNorm -> bf16 --------------------------
__global__ __launch_bounds__(256) void k_ln(const float* __restrict__ hin,
                                            const float* __restrict__ gam,
                                            const float* __restrict__ bet,
                                            __bf16* __restrict__ xbf) {
  const int row = blockIdx.x;
  const int tid = threadIdx.x;
  const float* hr = hin + (size_t)row * 2048;
  float v[8];
  float s = 0.f, ss = 0.f;
#pragma unroll
  for (int i = 0; i < 8; ++i) {
    float x = hr[tid + i * 256];
    v[i] = x; s += x; ss += x * x;
  }
#pragma unroll
  for (int o = 16; o > 0; o >>= 1) {
    s  += __shfl_xor(s,  o, 32);
    ss += __shfl_xor(ss, o, 32);
  }
  __shared__ float rs[8], rss[8];
  const int wave = tid >> 5, lane = tid & 31;
  if (lane == 0) { rs[wave] = s; rss[wave] = ss; }
  __syncthreads();
  float ts = 0.f, tss = 0.f;
#pragma unroll
  for (int i = 0; i < 8; ++i) { ts += rs[i]; tss += rss[i]; }
  const float mean = ts * (1.f / 2048.f);
  const float var  = tss * (1.f / 2048.f) - mean * mean;
  const float rstd = rsqrtf(var + 1e-5f);
  __bf16* xo = xbf + (size_t)row * 2048;
#pragma unroll
  for (int i = 0; i < 8; ++i) {
    int c = tid + i * 256;
    xo[c] = (__bf16)((v[i] - mean) * rstd * gam[c] + bet[c]);
  }
}

// ------------------------------- bf16 GEMM: out = A @ W^T + bias (+res) -----
// A: [M,K] bf16 row-major, W: [N,K] bf16 row-major, out: [M,N] fp32.
// 128x128 workgroup tile, 8 waves of 32, each wave 32x64 (2x4 WMMA frags).
// Double-buffered LDS staging via global_load_async_to_lds_b128: tile k+1 is
// DMA'd while tile k feeds the WMMA pipes (ASYNCcnt completes in order, so
// s_wait_asynccnt 4 == "tile k landed, tile k+1 still in flight").
#define LDS_STRIDE 40   // 32 k-elems + 8 pad: 20-dword row stride, bank-spread
__global__ __launch_bounds__(256) void k_gemm(const __bf16* __restrict__ A,
                                              const __bf16* __restrict__ W,
                                              const float* __restrict__ bias,
                                              const float* __restrict__ residual,
                                              float* __restrict__ out,
                                              int M, int N, int K) {
  __shared__ __align__(16) __bf16 sA[2][128 * LDS_STRIDE];
  __shared__ __align__(16) __bf16 sB[2][128 * LDS_STRIDE];
  const int tid  = threadIdx.x;
  const int lane = tid & 31;
  const int wave = tid >> 5;
  const int wm   = wave >> 1;          // 0..3  (M sub-blocks of 32)
  const int wn   = wave & 1;           // 0..1  (N sub-blocks of 64)
  const int M0   = blockIdx.y * 128;
  const int N0   = blockIdx.x * 128;
  const int lrow = lane & 15;
  const int hi   = lane >> 4;          // lane half selects K sub-range

  v8f acc[2][4];
#pragma unroll
  for (int i = 0; i < 2; ++i)
#pragma unroll
    for (int j = 0; j < 4; ++j) acc[i][j] = vzero8();

  const int ldrow = tid >> 1;
  const int ldcol = (tid & 1) << 4;
  const int lidx  = ldrow * LDS_STRIDE + ldcol;
  const __bf16* gA0 = A + (size_t)(M0 + ldrow) * K + ldcol;
  const __bf16* gB0 = W + (size_t)(N0 + ldrow) * K + ldcol;

  // stage tile 0 into buffer 0
  async_copy_b128(gA0,     &sA[0][lidx]);
  async_copy_b128(gA0 + 8, &sA[0][lidx + 8]);
  async_copy_b128(gB0,     &sB[0][lidx]);
  async_copy_b128(gB0 + 8, &sB[0][lidx + 8]);

  const int nk = K >> 5;
  for (int kt = 0; kt < nk; ++kt) {
    const int cur = kt & 1;
    if (kt + 1 < nk) {
      // DMA next tile into the other buffer (freed by last iter's barrier)
      const __bf16* gA = gA0 + (kt + 1) * 32;
      const __bf16* gB = gB0 + (kt + 1) * 32;
      async_copy_b128(gA,     &sA[cur ^ 1][lidx]);
      async_copy_b128(gA + 8, &sA[cur ^ 1][lidx + 8]);
      async_copy_b128(gB,     &sB[cur ^ 1][lidx]);
      async_copy_b128(gB + 8, &sB[cur ^ 1][lidx + 8]);
      __builtin_prefetch(gA + 32, 0, 3);   // tile k+2 toward GL2
      __builtin_prefetch(gB + 32, 0, 3);
      wait_async4();                       // tile k complete, k+1 in flight
    } else {
      wait_async0();
    }
    __syncthreads();

    FragBF af[2];
#pragma unroll
    for (int i = 0; i < 2; ++i) {
      const __bf16* p = &sA[cur][(wm * 32 + i * 16 + lrow) * LDS_STRIDE];
      af[i].q[0] = *(const u32x4*)(p + (hi ? 8 : 0));        // K 0..7 / 8..15
      af[i].q[1] = *(const u32x4*)(p + 16 + (hi ? 8 : 0));   // K 16..23 / 24..31
    }
    FragBF bfr[4];
#pragma unroll
    for (int j = 0; j < 4; ++j) {
      const __bf16* p = &sB[cur][(wn * 64 + j * 16 + lrow) * LDS_STRIDE];
      bfr[j].q[0] = *(const u32x4*)(p + (hi ? 8 : 0));       // K 0..7 / 8..15
      bfr[j].q[1] = *(const u32x4*)(p + 16 + (hi ? 8 : 0));  // K 16..23 / 24..31
    }
#pragma unroll
    for (int i = 0; i < 2; ++i)
#pragma unroll
      for (int j = 0; j < 4; ++j)
        acc[i][j] = wmma_bf16(af[i], bfr[j], acc[i][j]);
    __syncthreads();                       // all reads done -> buffer reusable
  }

  const bool hasres = (residual != nullptr);
#pragma unroll
  for (int i = 0; i < 2; ++i)
#pragma unroll
    for (int j = 0; j < 4; ++j) {
      const int n = N0 + wn * 64 + j * 16 + lrow;
      const float bv = bias[n];
#pragma unroll
      for (int r = 0; r < 8; ++r) {
        const int m = M0 + wm * 32 + i * 16 + r + hi * 8;
        float val = acc[i][j][r] + bv;
        if (hasres) val += residual[(size_t)m * N + n];
        out[(size_t)m * N + n] = val;
      }
    }
}

// ------------------------------- RoPE + pack to bf16 ------------------------
// q pre-scaled by 1/sqrt(d_k) so attention S = Q K^T needs no extra scale.
__global__ __launch_bounds__(256) void k_rope(const float* __restrict__ qf,
                                              const float* __restrict__ kf,
                                              const float* __restrict__ vf,
                                              __bf16* __restrict__ qo,
                                              __bf16* __restrict__ ko,
                                              __bf16* __restrict__ vo) {
  const int t   = blockIdx.x * 256 + threadIdx.x;   // 4096*16*64 threads
  const int i   = t & 63;
  const int hh  = (t >> 6) & 15;
  const int row = t >> 10;
  const int pos = row & 2047;
  const float theta = __powf(10000.f, -(float)i * (1.f / 64.f));
  const float f = (float)pos * theta;
  float sn, cs;
  __sincosf(f, &sn, &cs);
  const size_t base = ((size_t)row * 16 + hh) * 128;
  const float qscale = 0.08838834764831845f;        // 1/sqrt(128)
  const float q1 = qf[base + i], q2 = qf[base + i + 64];
  qo[base + i]      = (__bf16)((q1 * cs - q2 * sn) * qscale);
  qo[base + i + 64] = (__bf16)((q2 * cs + q1 * sn) * qscale);
  const float k1 = kf[base + i], k2 = kf[base + i + 64];
  ko[base + i]      = (__bf16)(k1 * cs - k2 * sn);
  ko[base + i + 64] = (__bf16)(k2 * cs + k1 * sn);
  vo[base + i]      = (__bf16)vf[base + i];
  vo[base + i + 64] = (__bf16)vf[base + i + 64];
}

// ------------------------------- flash attention ----------------------------
// grid (32, 16, 2) = (query-block of 64, head, batch); 4 waves per WG,
// each wave owns one 16-row query tile; K/V staged in shared LDS.
// K tile staged with global_load_async_to_lds_b128 (ASYNCcnt DMA path);
// V tile needs a transpose, so it is staged with b128 global loads + scatter.
__global__ __launch_bounds__(128) void k_flash(const __bf16* __restrict__ Q,
                                               const __bf16* __restrict__ Kb,
                                               const __bf16* __restrict__ Vb,
                                               __bf16* __restrict__ O) {
  const int Q0   = blockIdx.x * 64;
  const int hh   = blockIdx.y;
  const int bb   = blockIdx.z;
  const int tid  = threadIdx.x;
  const int lane = tid & 31;
  const int wave = tid >> 5;
  const int lrow = lane & 15;
  const int hi   = lane >> 4;
  const int q0   = Q0 + wave * 16;

  __shared__ __align__(16) __bf16 sK[32 * 136];    // [key][d], 272B rows
  __shared__ __align__(16) __bf16 sVT[128 * 40];   // [d][key], 80B rows
  __shared__ float sP[4][16 * 33];                 // per-wave P transpose pad

  // Q fragments straight from global (A-layout per-lane byte pattern)
  FragBF qfr[4];
  const size_t qbase = ((size_t)(bb * 2048 + q0 + lrow) * 16 + hh) * 128;
#pragma unroll
  for (int c = 0; c < 4; ++c) {
    qfr[c].q[0] = *(const u32x4*)(Q + qbase + c * 32 + (hi ? 8 : 0));
    qfr[c].q[1] = *(const u32x4*)(Q + qbase + c * 32 + 16 + (hi ? 8 : 0));
  }

  v8f oacc[8];
#pragma unroll
  for (int f = 0; f < 8; ++f) oacc[f] = vzero8();
  v8f mst, lsum;
#pragma unroll
  for (int r = 0; r < 8; ++r) { mst[r] = -1e30f; lsum[r] = 0.f; }

  const int nj = (Q0 >> 5) + 2;                    // causal KV tile bound (WG)
  for (int jt = 0; jt < nj; ++jt) {
    const int j0 = jt * 32;

    // ---- K tile: 32 rows x 256B, async DMA, 512 x 16B chunks, 4/thread ----
#pragma unroll
    for (int c = 0; c < 4; ++c) {
      const int chunk = c * 128 + tid;             // 0..511
      const int key   = chunk >> 4;
      const int part  = chunk & 15;
      const __bf16* g =
          Kb + ((size_t)(bb * 2048 + j0 + key) * 16 + hh) * 128 + part * 8;
      async_copy_b128(g, &sK[key * 136 + part * 8]);
    }
    // ---- V tile: b128 global loads, scalar transpose scatter to LDS ----
#pragma unroll
    for (int c = 0; c < 4; ++c) {
      const int chunk = c * 128 + tid;
      const int key   = chunk >> 4;
      const int part  = chunk & 15;
      u32x4 d4 = *(const u32x4*)(
          Vb + ((size_t)(bb * 2048 + j0 + key) * 16 + hh) * 128 + part * 8);
      const __bf16* e = (const __bf16*)&d4;
#pragma unroll
      for (int x = 0; x < 8; ++x)
        sVT[(part * 8 + x) * 40 + key] = e[x];
    }
    wait_async0();                                 // K DMA complete (this wave)
    __syncthreads();                               // all waves staged

    // ---- S = Q K^T : two 16x16 f32 frags (keys 0..15 / 16..31) ----
    v8f s0 = vzero8(), s1 = vzero8();
#pragma unroll
    for (int c = 0; c < 4; ++c) {
      FragBF b0, b1;
      const __bf16* p0 = &sK[lrow * 136 + c * 32];
      b0.q[0] = *(const u32x4*)(p0 + (hi ? 8 : 0));
      b0.q[1] = *(const u32x4*)(p0 + 16 + (hi ? 8 : 0));
      const __bf16* p1 = &sK[(16 + lrow) * 136 + c * 32];
      b1.q[0] = *(const u32x4*)(p1 + (hi ? 8 : 0));
      b1.q[1] = *(const u32x4*)(p1 + 16 + (hi ? 8 : 0));
      s0 = wmma_bf16(qfr[c], b0, s0);
      s1 = wmma_bf16(qfr[c], b1, s1);
    }

    // ---- causal mask ----
#pragma unroll
    for (int r = 0; r < 8; ++r) {
      const int qrow = q0 + r + hi * 8;
      const int col  = j0 + lrow;
      if (col > qrow)      s0[r] = -1e30f;
      if (col + 16 > qrow) s1[r] = -1e30f;
    }

    // ---- online softmax (row r in lanes 0-15, row r+8 in lanes 16-31) ----
    v8f t;
#pragma unroll
    for (int r = 0; r < 8; ++r) t[r] = fmaxf(s0[r], s1[r]);
#pragma unroll
    for (int off = 1; off < 16; off <<= 1)
#pragma unroll
      for (int r = 0; r < 8; ++r) t[r] = fmaxf(t[r], __shfl_xor(t[r], off, 32));
    v8f mnew, sco;
#pragma unroll
    for (int r = 0; r < 8; ++r) {
      mnew[r] = fmaxf(mst[r], t[r]);
      sco[r]  = __expf(mst[r] - mnew[r]);
      s0[r]   = __expf(s0[r] - mnew[r]);
      s1[r]   = __expf(s1[r] - mnew[r]);
    }
    v8f rsum;
#pragma unroll
    for (int r = 0; r < 8; ++r) rsum[r] = s0[r] + s1[r];
#pragma unroll
    for (int off = 1; off < 16; off <<= 1)
#pragma unroll
      for (int r = 0; r < 8; ++r) rsum[r] += __shfl_xor(rsum[r], off, 32);
#pragma unroll
    for (int r = 0; r < 8; ++r) {
      lsum[r] = lsum[r] * sco[r] + rsum[r];
      mst[r]  = mnew[r];
    }
#pragma unroll
    for (int f = 0; f < 8; ++f)
#pragma unroll
      for (int r = 0; r < 8; ++r) oacc[f][r] *= sco[r];

    // ---- transpose P (C-layout) -> A-layout bf16 via per-wave LDS ----
    float* pw = &sP[wave][0];
#pragma unroll
    for (int r = 0; r < 8; ++r) {
      pw[(r + hi * 8) * 33 + lrow]      = s0[r];
      pw[(r + hi * 8) * 33 + 16 + lrow] = s1[r];
    }
    __syncthreads();                     // uniform across WG; also DS in-order
    FragBF pf;
#pragma unroll
    for (int e = 0; e < 16; ++e) {
      const int k = (e < 8 ? e : e + 8) + (hi ? 8 : 0);
      pf.v[e] = (__bf16)pw[lrow * 33 + k];
    }

    // ---- O += P V ----
#pragma unroll
    for (int nb = 0; nb < 8; ++nb) {
      FragBF vb;
      const __bf16* p = &sVT[(nb * 16 + lrow) * 40];
      vb.q[0] = *(const u32x4*)(p + (hi ? 8 : 0));
      vb.q[1] = *(const u32x4*)(p + 16 + (hi ? 8 : 0));
      oacc[nb] = wmma_bf16(pf, vb, oacc[nb]);
    }
    __syncthreads();                     // before next K/V tile overwrites LDS
  }

  // ---- normalize and store o (bf16, layout [b*s][h*128+d]) ----
  v8f rl;
#pragma unroll
  for (int r = 0; r < 8; ++r) rl[r] = 1.f / lsum[r];
#pragma unroll
  for (int nb = 0; nb < 8; ++nb)
#pragma unroll
    for (int r = 0; r < 8; ++r) {
      const size_t m = (size_t)(bb * 2048 + q0 + r + hi * 8);
      O[m * 2048 + hh * 128 + nb * 16 + lrow] = (__bf16)(oacc[nb][r] * rl[r]);
    }
}

// ---------------------------------------------------------------------------
extern "C" void kernel_launch(void* const* d_in, const int* in_sizes, int n_in,
                              void* d_out, int out_size, void* d_ws, size_t ws_size,
                              hipStream_t stream) {
  const float* h   = (const float*)d_in[0];
  const float* wq  = (const float*)d_in[1];
  const float* bq  = (const float*)d_in[2];
  const float* wk  = (const float*)d_in[3];
  const float* bk  = (const float*)d_in[4];
  const float* wv  = (const float*)d_in[5];
  const float* bv  = (const float*)d_in[6];
  const float* wo  = (const float*)d_in[7];
  const float* bo  = (const float*)d_in[8];
  const float* lng = (const float*)d_in[9];
  const float* lnb = (const float*)d_in[10];

  char* ws = (char*)d_ws;
  const size_t MB = 1024ull * 1024ull;
  __bf16* x_bf  = (__bf16*)(ws + 0);          // 16MB; reused as o_bf later
  __bf16* wq_bf = (__bf16*)(ws + 16 * MB);
  __bf16* wk_bf = (__bf16*)(ws + 24 * MB);
  __bf16* wv_bf = (__bf16*)(ws + 32 * MB);
  __bf16* wo_bf = (__bf16*)(ws + 40 * MB);
  float*  q_f   = (float*)(ws + 48 * MB);     // 32MB each
  float*  k_f   = (float*)(ws + 80 * MB);
  float*  v_f   = (float*)(ws + 112 * MB);
  __bf16* q_bf  = (__bf16*)(ws + 144 * MB);   // 16MB each
  __bf16* k_bf  = (__bf16*)(ws + 160 * MB);
  __bf16* v_bf  = (__bf16*)(ws + 176 * MB);
  __bf16* o_bf  = x_bf;                       // x no longer needed post-QKV

  // weights fp32 -> bf16 (2048*2048 = 4M elements each)
  k_cvt<<<2048, 256, 0, stream>>>(wq, wq_bf);
  k_cvt<<<2048, 256, 0, stream>>>(wk, wk_bf);
  k_cvt<<<2048, 256, 0, stream>>>(wv, wv_bf);
  k_cvt<<<2048, 256, 0, stream>>>(wo, wo_bf);

  // LayerNorm (4096 rows)
  k_ln<<<4096, 256, 0, stream>>>(h, lng, lnb, x_bf);

  // QKV projections
  dim3 gg(16, 32);   // N/128, M/128
  k_gemm<<<gg, 256, 0, stream>>>(x_bf, wq_bf, bq, nullptr, q_f, 4096, 2048, 2048);
  k_gemm<<<gg, 256, 0, stream>>>(x_bf, wk_bf, bk, nullptr, k_f, 4096, 2048, 2048);
  k_gemm<<<gg, 256, 0, stream>>>(x_bf, wv_bf, bv, nullptr, v_f, 4096, 2048, 2048);

  // RoPE + pack (4096*16*64 threads)
  k_rope<<<16384, 256, 0, stream>>>(q_f, k_f, v_f, q_bf, k_bf, v_bf);

  // causal flash attention
  k_flash<<<dim3(32, 16, 2), 128, 0, stream>>>(q_bf, k_bf, v_bf, o_bf);

  // output projection + bias + residual -> fp32 d_out
  k_gemm<<<gg, 256, 0, stream>>>(o_bf, wo_bf, bo, h, (float*)d_out, 4096, 2048, 2048);
}
